// TMGNN_26276609917013
// MI455X (gfx1250) — matrix operand order repaired and checked by reading, not compiled, not run on hardware
//
#include <hip/hip_runtime.h>
#include <hip/hip_bf16.h>

// ---------------------------------------------------------------------------
// TMGNN forward for MI455X (gfx1250), wave32, bf16 WMMA GEMMs.
// All dense GEMMs use v_wmma_f32_16x16x32_bf16; scatter/pool stages are fp32.
// Gumbel noise is not bit-replicable (JAX threefry) -> deterministic argmax.
// ---------------------------------------------------------------------------

#define N_NODES   8192
#define NFEAT     200
#define NFEAT_PAD 224
#define NHID      128
#define E_EDGES   131072
#define D_REP     384
#define CAT_W     712
#define CAT_PAD   736

typedef __attribute__((ext_vector_type(16))) __bf16 v16bf;
typedef __attribute__((ext_vector_type(8)))  float  v8f;

union FragU { v16bf v; uint4 q[2]; };

__device__ __forceinline__ unsigned short f2bf(float f) {
  unsigned u = __float_as_uint(f);
  unsigned r = u + 0x7fffu + ((u >> 16) & 1u);   // round-to-nearest-even
  return (unsigned short)(r >> 16);
}

__device__ __forceinline__ float sigmf(float x) { return 1.0f / (1.0f + expf(-x)); }

// A fragment: 16x32 bf16, row-major source, lda multiple of 8.
// Per ISA table: lane L (m=L&15, h=L>>4): elems 0..7 = K[8h..8h+7], elems 8..15 = K[16+8h..16+8h+7].
__device__ __forceinline__ v16bf load_a_frag(const unsigned short* __restrict__ A, int lda,
                                             int row0, int k0, int lane) {
  const int m = lane & 15, h = lane >> 4;
  const unsigned short* p = A + (size_t)(row0 + m) * lda + k0 + h * 8;
  FragU f;
  f.q[0] = *reinterpret_cast<const uint4*>(p);
  f.q[1] = *reinterpret_cast<const uint4*>(p + 16);
  return f.v;
}

// B fragment: 32x16 bf16 (KxN). Source is transposed weight Wt[M][K_pad] (row-major),
// so per lane (n=L&15, h=L>>4): elems 0..7 = K[8h..8h+7], elems 8..15 = K[16+8h..16+8h+7].
__device__ __forceinline__ v16bf load_b_frag(const unsigned short* __restrict__ Bt, int ldb,
                                             int col0, int k0, int lane) {
  const int n = lane & 15, h = lane >> 4;
  const unsigned short* p = Bt + (size_t)(col0 + n) * ldb + k0 + h * 8;
  FragU f;
  f.q[0] = *reinterpret_cast<const uint4*>(p);
  f.q[1] = *reinterpret_cast<const uint4*>(p + 16);
  return f.v;
}

__device__ __forceinline__ v8f wmma_bf16(v16bf a, v16bf b, v8f c) {
  return __builtin_amdgcn_wmma_f32_16x16x32_bf16(false, a, false, b, (short)0, c, false, false);
}

// C/D layout: lane L (n=L&15, h=L>>4), VGPR r -> row m = r + 8h.
template<int ACT>  // 0=none, 1=relu
__device__ __forceinline__ void store_acc(v8f acc, int row0, int col0, int lane,
                                          const float* __restrict__ bias,
                                          unsigned short* __restrict__ Obf, int ldob,
                                          float* __restrict__ Of, int ldof) {
  const int n = lane & 15, h = lane >> 4;
  const int col = col0 + n;
  const float bv = bias ? bias[col] : 0.0f;
#pragma unroll
  for (int r = 0; r < 8; ++r) {
    const int row = row0 + h * 8 + r;
    float v = acc[r] + bv;
    if (ACT == 1) v = v > 0.0f ? v : 0.0f;
    if (Of)  Of[(size_t)row * ldof + col] = v;
    if (Obf) Obf[(size_t)row * ldob + col] = f2bf(v);
  }
}

// 8 waves/block; block tile 128(M) x 64(N); wave tile 32x32 = 4 WMMAs per K-step.
template<int ACT>
__global__ void __launch_bounds__(256)
gemm_bf16(const unsigned short* __restrict__ A, int lda,
          const unsigned short* __restrict__ Bt, int ldb,
          const float* __restrict__ bias,
          unsigned short* __restrict__ Obf, int ldob,
          float* __restrict__ Of, int ldof, int K)
{
  const int tid  = threadIdx.x;
  const int wave = tid >> 5;
  const int lane = tid & 31;
  const int row0 = blockIdx.x * 128 + (wave >> 1) * 32;
  const int col0 = blockIdx.y * 64  + (wave & 1) * 32;
  v8f a00 = {0.f,0.f,0.f,0.f,0.f,0.f,0.f,0.f};
  v8f a01 = a00, a10 = a00, a11 = a00;
  for (int k = 0; k < K; k += 32) {
    v16bf fa0 = load_a_frag(A, lda, row0,      k, lane);
    v16bf fa1 = load_a_frag(A, lda, row0 + 16, k, lane);
    v16bf fb0 = load_b_frag(Bt, ldb, col0,      k, lane);
    v16bf fb1 = load_b_frag(Bt, ldb, col0 + 16, k, lane);
    a00 = wmma_bf16(fa0, fb0, a00);
    a01 = wmma_bf16(fa0, fb1, a01);
    a10 = wmma_bf16(fa1, fb0, a10);
    a11 = wmma_bf16(fa1, fb1, a11);
  }
  store_acc<ACT>(a00, row0,      col0,      lane, bias, Obf, ldob, Of, ldof);
  store_acc<ACT>(a01, row0,      col0 + 16, lane, bias, Obf, ldob, Of, ldof);
  store_acc<ACT>(a10, row0 + 16, col0,      lane, bias, Obf, ldob, Of, ldof);
  store_acc<ACT>(a11, row0 + 16, col0 + 16, lane, bias, Obf, ldob, Of, ldof);
}

// ------------------------- small / glue kernels ----------------------------

__global__ void scatter_mp(const float* __restrict__ X, const int* __restrict__ src,
                           const int* __restrict__ dst, const float* __restrict__ w,
                           float* __restrict__ out, int F, int total)
{
  int i = blockIdx.x * blockDim.x + threadIdx.x;
  if (i >= total) return;
  int e = i / F, f = i - e * F;
  atomicAdd(&out[(size_t)dst[e] * F + f], X[(size_t)src[e] * F + f] * w[e]);
}

__global__ void cvt_pad_bf16(const float* __restrict__ X, int F, int Fp,
                             unsigned short* __restrict__ Y, int total)
{
  int i = blockIdx.x * blockDim.x + threadIdx.x;
  if (i >= total) return;
  int n = i / Fp, c = i - n * Fp;
  Y[i] = (c < F) ? f2bf(X[(size_t)n * F + c]) : (unsigned short)0;
}

// W[K][M] fp32 -> Wt[M][Kp] bf16 (zero padded K)
__global__ void pack_wt(const float* __restrict__ W, int K, int M, int Kp,
                        unsigned short* __restrict__ Wt)
{
  int i = blockIdx.x * blockDim.x + threadIdx.x;
  if (i >= M * Kp) return;
  int n = i / Kp, k = i - n * Kp;
  Wt[i] = (k < K) ? f2bf(W[(size_t)k * M + n]) : (unsigned short)0;
}

__global__ void add_vec(const float* __restrict__ a, const float* __restrict__ b,
                        float* __restrict__ o, int n)
{
  int i = blockIdx.x * blockDim.x + threadIdx.x;
  if (i < n) o[i] = a[i] + b[i];
}

// logits = bottom @ ml_W0 + b0 ; c0 = argmax  (Gumbel noise omitted: deterministic)
__global__ void assign0_kernel(const float* __restrict__ bottom, const float* __restrict__ W,
                               const float* __restrict__ B, int* __restrict__ c0)
{
  int n = blockIdx.x * blockDim.x + threadIdx.x;
  if (n >= N_NODES) return;
  float s[5];
#pragma unroll
  for (int j = 0; j < 5; ++j) s[j] = B[j];
  const float* h = bottom + (size_t)n * NHID;
  for (int k = 0; k < NHID; ++k) {
    float hv = h[k];
#pragma unroll
    for (int j = 0; j < 5; ++j) s[j] += hv * W[k * 5 + j];
  }
  float best = s[0]; int bi = 0;
#pragma unroll
  for (int j = 1; j < 5; ++j) if (s[j] > best) { best = s[j]; bi = j; }
  c0[n] = bi;
}

__global__ void accum_xc0(const float* __restrict__ bottom, const int* __restrict__ c0,
                          float* __restrict__ xc0)
{
  int i = blockIdx.x * blockDim.x + threadIdx.x;
  if (i >= N_NODES * NHID) return;
  int n = i >> 7, f = i & 127;
  atomicAdd(&xc0[c0[n] * NHID + f], bottom[i]);
}

__global__ void adj_accum(const int* __restrict__ src, const int* __restrict__ dst,
                          const int* __restrict__ c0, float* __restrict__ adj1)
{
  int e = blockIdx.x * blockDim.x + threadIdx.x;
  if (e >= E_EDGES) return;
  atomicAdd(&adj1[c0[src[e]] * 5 + c0[dst[e]]], 1.0f);
}

// Entire coarse (cluster-level) pipeline in one block: tiny matrices.
__global__ void __launch_bounds__(256)
coarse_pool(const float* __restrict__ xc0_raw, const float* __restrict__ adj1_raw,
            const float* __restrict__ meW0, const float* __restrict__ meB0,
            const float* __restrict__ mlW1, const float* __restrict__ mlB1,
            const float* __restrict__ meW1, const float* __restrict__ meB1,
            float* __restrict__ latent0, float* __restrict__ latent1,
            int* __restrict__ c1_out)
{
  __shared__ float s_xc0[5][128];
  __shared__ float s_adj1[5][5];
  __shared__ float s_t0[5][128];
  __shared__ float s_l0[5][128];
  __shared__ float s_xc1[10][128];
  __shared__ float s_adj2[10][10];
  __shared__ float s_t1[10][128];
  __shared__ int   s_c1[5];
  __shared__ float s_red[64];
  const int tid = threadIdx.x;

  for (int i = tid; i < 5 * 128; i += 256) s_xc0[i / 128][i % 128] = xc0_raw[i];
  if (tid < 25) s_adj1[tid / 5][tid % 5] = adj1_raw[tid];
  __syncthreads();
  if (tid < 5) {
    float ss = 0.f;
    for (int k = 0; k < 128; ++k) { float v = s_xc0[tid][k]; ss += v * v; }
    s_red[tid] = 1.0f / fmaxf(sqrtf(ss), 1e-12f);
  }
  if (tid == 0) {
    float S = 0.f;
    for (int i = 0; i < 25; ++i) S += adj1_raw[i];
    s_red[8] = 1.0f / S;
  }
  __syncthreads();
  for (int i = tid; i < 5 * 128; i += 256) s_xc0[i / 128][i % 128] *= s_red[i / 128];
  if (tid < 25) s_adj1[tid / 5][tid % 5] *= s_red[8];
  __syncthreads();
  // t0 = adj1n @ xc0n
  for (int i = tid; i < 5 * 128; i += 256) {
    int r = i / 128, f = i % 128;
    float acc = 0.f;
    for (int j = 0; j < 5; ++j) acc += s_adj1[r][j] * s_xc0[j][f];
    s_t0[r][f] = acc;
  }
  __syncthreads();
  // latent0 = tanh(t0 @ meW0 + b0)
  for (int i = tid; i < 5 * 128; i += 256) {
    int r = i / 128, o = i % 128;
    float acc = meB0[o];
    for (int k = 0; k < 128; ++k) acc += s_t0[r][k] * meW0[k * 128 + o];
    s_l0[r][o] = tanhf(acc);
  }
  __syncthreads();
  for (int i = tid; i < 5 * 128; i += 256) latent0[i] = s_l0[i / 128][i % 128];
  // level-1 assignment (argmax of latent0 @ mlW1 + b1)
  if (tid < 5) {
    float best = -3.4e38f; int bi = 0;
    for (int j = 0; j < 10; ++j) {
      float s = mlB1[j];
      for (int k = 0; k < 128; ++k) s += s_l0[tid][k] * mlW1[k * 10 + j];
      if (s > best) { best = s; bi = j; }
    }
    s_c1[tid] = bi;
    c1_out[tid] = bi;
  }
  __syncthreads();
  // xc1 = assign1^T @ latent0 ; adj2 = assign1^T @ adj1n @ assign1
  for (int i = tid; i < 10 * 128; i += 256) {
    int j = i / 128, f = i % 128;
    float acc = 0.f;
    for (int k = 0; k < 5; ++k) if (s_c1[k] == j) acc += s_l0[k][f];
    s_xc1[j][f] = acc;
  }
  if (tid < 100) {
    int a = tid / 10, b = tid % 10;
    float acc = 0.f;
    for (int ii = 0; ii < 5; ++ii)
      for (int jj = 0; jj < 5; ++jj)
        if (s_c1[ii] == a && s_c1[jj] == b) acc += s_adj1[ii][jj];
    s_adj2[a][b] = acc;
  }
  __syncthreads();
  if (tid < 10) {
    float ss = 0.f;
    for (int k = 0; k < 128; ++k) { float v = s_xc1[tid][k]; ss += v * v; }
    s_red[16 + tid] = 1.0f / fmaxf(sqrtf(ss), 1e-12f);
  }
  if (tid == 0) {
    float S = 0.f;
    for (int i = 0; i < 100; ++i) S += s_adj2[i / 10][i % 10];
    s_red[40] = 1.0f / S;
  }
  __syncthreads();
  for (int i = tid; i < 10 * 128; i += 256) s_xc1[i / 128][i % 128] *= s_red[16 + i / 128];
  if (tid < 100) s_adj2[tid / 10][tid % 10] *= s_red[40];
  __syncthreads();
  for (int i = tid; i < 10 * 128; i += 256) {
    int r = i / 128, f = i % 128;
    float acc = 0.f;
    for (int j = 0; j < 10; ++j) acc += s_adj2[r][j] * s_xc1[j][f];
    s_t1[r][f] = acc;
  }
  __syncthreads();
  for (int i = tid; i < 10 * 128; i += 256) {
    int r = i / 128, o = i % 128;
    float acc = meB1[o];
    for (int k = 0; k < 128; ++k) acc += s_t1[r][k] * meW1[k * 128 + o];
    latent1[i] = tanhf(acc);
  }
}

// rep[:,128:256] = latent0[c0[n]] ; rep[:,256:384] = latent1[c1[c0[n]]]
__global__ void write_rep(const int* __restrict__ c0, const int* __restrict__ c1,
                          const float* __restrict__ lat0, const float* __restrict__ lat1,
                          unsigned short* __restrict__ rep)
{
  int i = blockIdx.x * blockDim.x + threadIdx.x;
  if (i >= N_NODES * NHID) return;
  int n = i >> 7, f = i & 127;
  int a = c0[n], b = c1[a];
  rep[(size_t)n * D_REP + 128 + f] = f2bf(lat0[a * NHID + f]);
  rep[(size_t)n * D_REP + 256 + f] = f2bf(lat1[b * NHID + f]);
}

// LSTM cell with h0=c0=0: gates order i,f,g,o; c=sig(i)*tanh(g); h=sig(o)*tanh(c)
__global__ void lstm_elem(const float* __restrict__ gf, const float* __restrict__ gb,
                          unsigned short* __restrict__ out)
{
  int i = blockIdx.x * blockDim.x + threadIdx.x;
  if (i >= N_NODES * NHID) return;
  int n = i >> 7, j = i & 127;
  const float* pf = gf + (size_t)n * 512;
  float cf = sigmf(pf[j]) * tanhf(pf[256 + j]);
  float hf = sigmf(pf[384 + j]) * tanhf(cf);
  const float* pb = gb + (size_t)n * 512;
  float cb = sigmf(pb[j]) * tanhf(pb[256 + j]);
  float hb = sigmf(pb[384 + j]) * tanhf(cb);
  out[(size_t)n * 256 + j]       = f2bf(hf);
  out[(size_t)n * 256 + 128 + j] = f2bf(hb);
}

__global__ void concat_cat(const unsigned short* __restrict__ o1,
                           const unsigned short* __restrict__ o2,
                           const float* __restrict__ x,
                           unsigned short* __restrict__ cat, int total)
{
  int i = blockIdx.x * blockDim.x + threadIdx.x;
  if (i >= total) return;
  int n = i / CAT_PAD, c = i - n * CAT_PAD;
  unsigned short v;
  if (c < 256)       v = o1[(size_t)n * 256 + c];
  else if (c < 512)  v = o2[(size_t)n * 256 + (c - 256)];
  else if (c < CAT_W) v = f2bf(x[(size_t)n * NFEAT + (c - 512)]);
  else v = 0;
  cat[i] = v;
}

__global__ void fc2_logsoftmax(const float* __restrict__ H, const float* __restrict__ W,
                               const float* __restrict__ B, float* __restrict__ out)
{
  int n = blockIdx.x * blockDim.x + threadIdx.x;
  if (n >= N_NODES) return;
  float z[16];
#pragma unroll
  for (int j = 0; j < 16; ++j) z[j] = B[j];
  const float* h = H + (size_t)n * NHID;
  for (int k = 0; k < NHID; ++k) {
    float hv = h[k];
#pragma unroll
    for (int j = 0; j < 16; ++j) z[j] += hv * W[k * 16 + j];
  }
  float m = z[0];
#pragma unroll
  for (int j = 1; j < 16; ++j) m = fmaxf(m, z[j]);
  float s = 0.f;
#pragma unroll
  for (int j = 0; j < 16; ++j) s += expf(z[j] - m);
  float l = logf(s);
#pragma unroll
  for (int j = 0; j < 16; ++j) out[(size_t)n * 16 + j] = z[j] - m - l;
}

// ---------------------------------------------------------------------------

extern "C" void kernel_launch(void* const* d_in, const int* in_sizes, int n_in,
                              void* d_out, int out_size, void* d_ws, size_t ws_size,
                              hipStream_t stream) {
  (void)in_sizes; (void)n_in; (void)out_size; (void)ws_size;
  const float* x        = (const float*)d_in[0];
  const int*   eidx     = (const int*)d_in[1];
  const float* eattr    = (const float*)d_in[2];
  const float* enc_W1   = (const float*)d_in[3];
  const float* enc_b1   = (const float*)d_in[4];
  const float* enc_W2   = (const float*)d_in[5];
  const float* enc_b2   = (const float*)d_in[6];
  const float* ml_W0    = (const float*)d_in[7];
  const float* ml_b0    = (const float*)d_in[8];
  const float* me_W0    = (const float*)d_in[9];
  const float* me_b0    = (const float*)d_in[10];
  const float* ml_W1    = (const float*)d_in[11];
  const float* ml_b1    = (const float*)d_in[12];
  const float* me_W1    = (const float*)d_in[13];
  const float* me_b1    = (const float*)d_in[14];
  const float* mix1_W   = (const float*)d_in[15];
  const float* mix1_b   = (const float*)d_in[16];
  const float* mix2_W   = (const float*)d_in[17];
  const float* mix2_b   = (const float*)d_in[18];
  const float* r1_Wih_f = (const float*)d_in[19];
  const float* r1_bih_f = (const float*)d_in[20];
  const float* r1_bhh_f = (const float*)d_in[22];   // Whh unused (h0=0)
  const float* r1_Wih_b = (const float*)d_in[23];
  const float* r1_bih_b = (const float*)d_in[24];
  const float* r1_bhh_b = (const float*)d_in[26];
  const float* r2_Wih_f = (const float*)d_in[27];
  const float* r2_bih_f = (const float*)d_in[28];
  const float* r2_bhh_f = (const float*)d_in[30];
  const float* r2_Wih_b = (const float*)d_in[31];
  const float* r2_bih_b = (const float*)d_in[32];
  const float* r2_bhh_b = (const float*)d_in[34];
  const float* fc1_W    = (const float*)d_in[35];
  const float* fc1_b    = (const float*)d_in[36];
  const float* fc2_W    = (const float*)d_in[37];
  const float* fc2_b    = (const float*)d_in[38];
  const int* src = eidx;
  const int* dst = eidx + E_EDGES;

  // ---- workspace layout ----
  char* ws = (char*)d_ws;
  size_t off = 0;
  auto alloc = [&](size_t bytes) -> void* {
    off = (off + 255) & ~(size_t)255;
    void* p = ws + off;
    off += bytes;
    return p;
  };
  // Union region: encoder chain / LSTM gates / concat buffer (lifetimes disjoint)
  char* uni = (char*)alloc(33554432);
  float*          mpx_f    = (float*)(uni + 0);            // N*200 f32
  unsigned short* mpx_bf   = (unsigned short*)(uni + 6553600);   // N*224 bf16
  float*          h_f      = (float*)(uni + 10223616);     // N*128 f32
  float*          mph_f    = (float*)(uni + 14417920);     // N*128 f32
  unsigned short* mph_bf   = (unsigned short*)(uni + 18612224);  // N*128 bf16
  float*          bottom_f = (float*)(uni + 20709376);     // N*128 f32
  float*          gatesF   = (float*)(uni + 0);            // N*512 f32
  float*          gatesB   = (float*)(uni + 16777216);     // N*512 f32
  unsigned short* cat_bf   = (unsigned short*)(uni + 0);   // N*736 bf16

  unsigned short* rep_bf = (unsigned short*)alloc((size_t)N_NODES * D_REP * 2);
  unsigned short* h512   = (unsigned short*)alloc((size_t)N_NODES * 512 * 2);
  unsigned short* h384   = (unsigned short*)alloc((size_t)N_NODES * 384 * 2);
  unsigned short* out1   = (unsigned short*)alloc((size_t)N_NODES * 256 * 2);
  unsigned short* out2   = (unsigned short*)alloc((size_t)N_NODES * 256 * 2);
  float*          fc1h   = (float*)alloc((size_t)N_NODES * NHID * 4);
  int*   c0   = (int*)alloc(N_NODES * 4);
  int*   c1   = (int*)alloc(64);
  float* xc0  = (float*)alloc(5 * 128 * 4);
  float* adj1 = (float*)alloc(25 * 4);
  float* lat0 = (float*)alloc(5 * 128 * 4);
  float* lat1 = (float*)alloc(10 * 128 * 4);
  float* bs_r1f = (float*)alloc(512 * 4);
  float* bs_r1b = (float*)alloc(512 * 4);
  float* bs_r2f = (float*)alloc(512 * 4);
  float* bs_r2b = (float*)alloc(512 * 4);
  unsigned short* Wt_enc1 = (unsigned short*)alloc(128 * 224 * 2);
  unsigned short* Wt_enc2 = (unsigned short*)alloc(128 * 128 * 2);
  unsigned short* Wt_mix1 = (unsigned short*)alloc(512 * 384 * 2);
  unsigned short* Wt_mix2 = (unsigned short*)alloc(384 * 512 * 2);
  unsigned short* Wt_r1f  = (unsigned short*)alloc(512 * 384 * 2);
  unsigned short* Wt_r1b  = (unsigned short*)alloc(512 * 384 * 2);
  unsigned short* Wt_r2f  = (unsigned short*)alloc(512 * 256 * 2);
  unsigned short* Wt_r2b  = (unsigned short*)alloc(512 * 256 * 2);
  unsigned short* Wt_fc1  = (unsigned short*)alloc(128 * 736 * 2);

  auto nblk = [](long long n) { return (unsigned)((n + 255) / 256); };
  const dim3 B256(256);

  // ---- zero accumulators (re-done every call; graph-capture safe) ----
  hipMemsetAsync(mpx_f, 0, (size_t)N_NODES * NFEAT * 4, stream);
  hipMemsetAsync(mph_f, 0, (size_t)N_NODES * NHID * 4, stream);
  hipMemsetAsync(xc0,  0, 5 * 128 * 4, stream);
  hipMemsetAsync(adj1, 0, 25 * 4, stream);

  // ---- pack weights (fp32 [K,M] -> bf16 Wt[M,Kp]) + fused LSTM biases ----
  pack_wt<<<nblk(128 * 224), B256, 0, stream>>>(enc_W1, 200, 128, 224, Wt_enc1);
  pack_wt<<<nblk(128 * 128), B256, 0, stream>>>(enc_W2, 128, 128, 128, Wt_enc2);
  pack_wt<<<nblk(512 * 384), B256, 0, stream>>>(mix1_W, 384, 512, 384, Wt_mix1);
  pack_wt<<<nblk(384 * 512), B256, 0, stream>>>(mix2_W, 512, 384, 512, Wt_mix2);
  pack_wt<<<nblk(512 * 384), B256, 0, stream>>>(r1_Wih_f, 384, 512, 384, Wt_r1f);
  pack_wt<<<nblk(512 * 384), B256, 0, stream>>>(r1_Wih_b, 384, 512, 384, Wt_r1b);
  pack_wt<<<nblk(512 * 256), B256, 0, stream>>>(r2_Wih_f, 256, 512, 256, Wt_r2f);
  pack_wt<<<nblk(512 * 256), B256, 0, stream>>>(r2_Wih_b, 256, 512, 256, Wt_r2b);
  pack_wt<<<nblk(128 * 736), B256, 0, stream>>>(fc1_W, 712, 128, 736, Wt_fc1);
  add_vec<<<2, B256, 0, stream>>>(r1_bih_f, r1_bhh_f, bs_r1f, 512);
  add_vec<<<2, B256, 0, stream>>>(r1_bih_b, r1_bhh_b, bs_r1b, 512);
  add_vec<<<2, B256, 0, stream>>>(r2_bih_f, r2_bhh_f, bs_r2f, 512);
  add_vec<<<2, B256, 0, stream>>>(r2_bih_b, r2_bhh_b, bs_r2b, 512);

  // ---- encoder: h = relu(mp(x)@W1+b1); bottom = mp(h)@W2+b2 ----
  scatter_mp<<<nblk((long long)E_EDGES * NFEAT), B256, 0, stream>>>(x, src, dst, eattr, mpx_f, NFEAT, E_EDGES * NFEAT);
  cvt_pad_bf16<<<nblk((long long)N_NODES * NFEAT_PAD), B256, 0, stream>>>(mpx_f, NFEAT, NFEAT_PAD, mpx_bf, N_NODES * NFEAT_PAD);
  gemm_bf16<1><<<dim3(64, 2), B256, 0, stream>>>(mpx_bf, NFEAT_PAD, Wt_enc1, NFEAT_PAD, enc_b1,
                                                 (unsigned short*)nullptr, 0, h_f, NHID, NFEAT_PAD);
  scatter_mp<<<nblk((long long)E_EDGES * NHID), B256, 0, stream>>>(h_f, src, dst, eattr, mph_f, NHID, E_EDGES * NHID);
  cvt_pad_bf16<<<nblk((long long)N_NODES * NHID), B256, 0, stream>>>(mph_f, NHID, NHID, mph_bf, N_NODES * NHID);
  // bottom: fp32 for pooling + bf16 straight into rep[:,0:128]
  gemm_bf16<0><<<dim3(64, 2), B256, 0, stream>>>(mph_bf, NHID, Wt_enc2, NHID, enc_b2,
                                                 rep_bf, D_REP, bottom_f, NHID, NHID);

  // ---- hierarchical pooling (hard assignments; N x N adj never materialized) ----
  assign0_kernel<<<nblk(N_NODES), B256, 0, stream>>>(bottom_f, ml_W0, ml_b0, c0);
  accum_xc0<<<nblk((long long)N_NODES * NHID), B256, 0, stream>>>(bottom_f, c0, xc0);
  adj_accum<<<nblk(E_EDGES), B256, 0, stream>>>(src, dst, c0, adj1);
  coarse_pool<<<1, B256, 0, stream>>>(xc0, adj1, me_W0, me_b0, ml_W1, ml_b1, me_W1, me_b1,
                                      lat0, lat1, c1);
  write_rep<<<nblk((long long)N_NODES * NHID), B256, 0, stream>>>(c0, c1, lat0, lat1, rep_bf);

  // ---- mixer ----
  gemm_bf16<1><<<dim3(64, 8), B256, 0, stream>>>(rep_bf, D_REP, Wt_mix1, D_REP, mix1_b,
                                                 h512, 512, (float*)nullptr, 0, D_REP);
  gemm_bf16<1><<<dim3(64, 6), B256, 0, stream>>>(h512, 512, Wt_mix2, 512, mix2_b,
                                                 h384, D_REP, (float*)nullptr, 0, 512);

  // ---- bi-LSTM 1 (seq_len=1, h0=c0=0 -> gates = X@Wih + (bih+bhh)) ----
  gemm_bf16<0><<<dim3(64, 8), B256, 0, stream>>>(h384, D_REP, Wt_r1f, D_REP, bs_r1f,
                                                 (unsigned short*)nullptr, 0, gatesF, 512, D_REP);
  gemm_bf16<0><<<dim3(64, 8), B256, 0, stream>>>(h384, D_REP, Wt_r1b, D_REP, bs_r1b,
                                                 (unsigned short*)nullptr, 0, gatesB, 512, D_REP);
  lstm_elem<<<nblk((long long)N_NODES * NHID), B256, 0, stream>>>(gatesF, gatesB, out1);

  // ---- bi-LSTM 2 ----
  gemm_bf16<0><<<dim3(64, 8), B256, 0, stream>>>(out1, 256, Wt_r2f, 256, bs_r2f,
                                                 (unsigned short*)nullptr, 0, gatesF, 512, 256);
  gemm_bf16<0><<<dim3(64, 8), B256, 0, stream>>>(out1, 256, Wt_r2b, 256, bs_r2b,
                                                 (unsigned short*)nullptr, 0, gatesB, 512, 256);
  lstm_elem<<<nblk((long long)N_NODES * NHID), B256, 0, stream>>>(gatesF, gatesB, out2);

  // ---- head: fc1 (concat) -> fc2 + log_softmax ----
  concat_cat<<<nblk((long long)N_NODES * CAT_PAD), B256, 0, stream>>>(out1, out2, x, cat_bf,
                                                                     N_NODES * CAT_PAD);
  gemm_bf16<1><<<dim3(64, 2), B256, 0, stream>>>(cat_bf, CAT_PAD, Wt_fc1, CAT_PAD, fc1_b,
                                                 (unsigned short*)nullptr, 0, fc1h, NHID, CAT_PAD);
  fc2_logsoftmax<<<nblk(N_NODES), B256, 0, stream>>>(fc1h, fc2_W, fc2_b, (float*)d_out);
}
